// HMM_83365315215750
// MI455X (gfx1250) — matrix operand order repaired and checked by reading, not compiled, not run on hardware
//
#include <hip/hip_runtime.h>

typedef __attribute__((ext_vector_type(16))) _Float16 v16h;
typedef __attribute__((ext_vector_type(8)))  float    v8f;

#define NS     4097            // states / symbols (incl. bookend)
#define SP     4160            // padded state dim (multiple of 64 -> static trip count)
#define TLEN   2048            // sequence length
#define NTILE  257             // ceil(4097/16) 16-wide output tiles
#define KSPLIT 2               // k-split waves per tile
#define KITER  (SP / (KSPLIT * 32))   // 65 compile-time iterations per half
#define NWAVE  (NTILE * KSPLIT)
#define BTHR   128
#define WPB    (BTHR / 32)
#define GBLK   ((NWAVE + WPB - 1) / WPB)   // 129 blocks

// ---------------------------------------------------------------------------
// Grid-wide barrier (all GBLK blocks resident; tiny footprint guarantees it).
// Monotonic generation counter -> no reset races; fences give release/acquire.
// ---------------------------------------------------------------------------
__device__ __forceinline__ void grid_barrier(unsigned* cnt, unsigned* rel, unsigned gen) {
    __threadfence();                 // release my stores to device scope
    __syncthreads();
    if (threadIdx.x == 0) {
        unsigned arrived = atomicAdd(cnt, 1u) + 1u;
        if (arrived == gen * (unsigned)GBLK) {
            atomicExch(rel, gen);    // last block releases this generation
        } else {
            while (atomicAdd(rel, 0u) < gen) { __builtin_amdgcn_s_sleep(2); }
        }
    }
    __syncthreads();
    __threadfence();                 // acquire: invalidate L0 before reading peers' data
}

// ---------------------------------------------------------------------------
// prep: transT[j][k] = (fp16) exp(log_trans[k][j]), zero-padded to SP x SP.
// LDS-tiled transpose so both the read and the write are coalesced.
// ---------------------------------------------------------------------------
__global__ void hmm_prep(const float* __restrict__ lt, _Float16* __restrict__ tT) {
    __shared__ float tile[32][33];
    const int kb = blockIdx.x * 32;
    const int jb = blockIdx.y * 32;
    for (int yy = threadIdx.y; yy < 32; yy += 8) {
        const int k = kb + yy, j = jb + threadIdx.x;
        tile[yy][threadIdx.x] = (k < NS && j < NS) ? lt[(size_t)k * NS + j] : -1.0e30f;
    }
    __syncthreads();
    for (int yy = threadIdx.y; yy < 32; yy += 8) {
        const int j = jb + yy, k = kb + threadIdx.x;
        tT[(size_t)j * SP + k] = (_Float16)__expf(tile[threadIdx.x][yy]);
    }
}

// ---------------------------------------------------------------------------
// init: fs0 = exp(log_pi) in partial buffer A, everything else zeroed.
// slotA[0] holds float 1.0 so the t=0 "previous max" is exactly 1.
// ---------------------------------------------------------------------------
__global__ void hmm_init(const float* __restrict__ log_pi,
                         float* yA0, float* yA1, float* yB0, float* yB1,
                         unsigned* slotA, unsigned* slotB, unsigned* bar) {
    const int i = blockIdx.x * blockDim.x + threadIdx.x;
    if (i < SP) {
        yA0[i] = (i < NS) ? __expf(log_pi[i]) : 0.0f;
        yA1[i] = 0.0f; yB0[i] = 0.0f; yB1[i] = 0.0f;
    }
    if (i < TLEN + 1) {
        slotA[i] = (i == 0) ? __float_as_uint(1.0f) : 0u;
        slotB[i] = 0u;
    }
    if (i == 0) { bar[0] = 0u; bar[1] = 0u; }
}

// ---------------------------------------------------------------------------
// One k-chunk of 32: build the broadcast-A fp16 fragment from the two fp32
// partial vectors (scaled by inv), load the contiguous B fragment, WMMA.
// ---------------------------------------------------------------------------
__device__ __forceinline__ v8f wmma_chunk(const float* aA, const float* aB,
                                          const _Float16* brow, int k0,
                                          float inv, v8f acc) {
    const float4 pa0 = *(const float4*)(aA + k0);
    const float4 pa1 = *(const float4*)(aA + k0 + 4);
    const float4 pa2 = *(const float4*)(aA + k0 + 8);
    const float4 pa3 = *(const float4*)(aA + k0 + 12);
    const float4 pb0 = *(const float4*)(aB + k0);
    const float4 pb1 = *(const float4*)(aB + k0 + 4);
    const float4 pb2 = *(const float4*)(aB + k0 + 8);
    const float4 pb3 = *(const float4*)(aB + k0 + 12);
    v16h a;
    a[0]  = (_Float16)((pa0.x + pb0.x) * inv);
    a[1]  = (_Float16)((pa0.y + pb0.y) * inv);
    a[2]  = (_Float16)((pa0.z + pb0.z) * inv);
    a[3]  = (_Float16)((pa0.w + pb0.w) * inv);
    a[4]  = (_Float16)((pa1.x + pb1.x) * inv);
    a[5]  = (_Float16)((pa1.y + pb1.y) * inv);
    a[6]  = (_Float16)((pa1.z + pb1.z) * inv);
    a[7]  = (_Float16)((pa1.w + pb1.w) * inv);
    a[8]  = (_Float16)((pa2.x + pb2.x) * inv);
    a[9]  = (_Float16)((pa2.y + pb2.y) * inv);
    a[10] = (_Float16)((pa2.z + pb2.z) * inv);
    a[11] = (_Float16)((pa2.w + pb2.w) * inv);
    a[12] = (_Float16)((pa3.x + pb3.x) * inv);
    a[13] = (_Float16)((pa3.y + pb3.y) * inv);
    a[14] = (_Float16)((pa3.z + pb3.z) * inv);
    a[15] = (_Float16)((pa3.w + pb3.w) * inv);
    const v16h b = *(const v16h*)(brow + k0);
    return __builtin_amdgcn_wmma_f32_16x16x32_f16(
               false, a, false, b, (short)0, acc, false, false);
}

// ---------------------------------------------------------------------------
// forward: persistent kernel. One WMMA matvec + one grid barrier per step.
//   y_t = transT * (y_{t-1}/m_{t-1}) .* em(:,obs_t)   (linear domain, fp32 acc)
//   L   = sum log(m_t)   ;   out = L + log(sum_j y_T[j] * trans[j][0])
// ---------------------------------------------------------------------------
__global__ void __launch_bounds__(BTHR) hmm_forward(
    const _Float16* __restrict__ tT,
    const float* __restrict__ log_em,
    const float* __restrict__ log_trans,
    const int*  __restrict__ obvs,
    float* __restrict__ yA0, float* __restrict__ yA1,
    float* __restrict__ yB0, float* __restrict__ yB1,
    unsigned* __restrict__ slotA, unsigned* __restrict__ slotB,
    unsigned* __restrict__ bar, float* __restrict__ out)
{
    const int lane = threadIdx.x & 31;
    const int w    = blockIdx.x * WPB + (threadIdx.x >> 5);
    const int tile = w >> 1;        // 16-wide output tile
    const int half = w & 1;         // k-split index
    const int l15  = lane & 15;
    const int hs   = lane >> 4;     // lane half-select (A/B fragment layout)
    unsigned* msA = slotA + 1;      // msX[-1] = slotX[0] (initial scale)
    unsigned* msB = slotB + 1;
    float* bufA[2] = { yA0, yA1 };
    float* bufB[2] = { yB0, yB1 };
    double L = 0.0;

    for (int t = 0; t < TLEN; ++t) {
        const float* rA = bufA[t & 1];
        const float* rB = bufB[t & 1];
        float*    wY = (half == 0) ? bufA[(t + 1) & 1] : bufB[(t + 1) & 1];
        unsigned* wS = (half == 0) ? (msA + t) : (msB + t);

        const float m_prev = __uint_as_float(msA[t - 1]) + __uint_as_float(msB[t - 1]);
        if (t >= 1 && threadIdx.x == 0 && blockIdx.x == 0) L += (double)logf(m_prev);
        const float inv = 1.0f / m_prev;
        const int   obs = obvs[t];

        if (tile < NTILE) {   // uniform per wave -> EXEC all ones for WMMA
            const int j0 = tile * 16;
            // B fragment: lane covers column j0+l15, K-halves picked by hs; contiguous 32B.
            const _Float16* brow = tT + (size_t)(j0 + l15) * SP + hs * 16;
            // A fragment (broadcast fs): lanes<16 need K {0..15}, lanes>=16 {16..31}.
            const float* aA = rA + hs * 16;
            const float* aB = rB + hs * 16;
            const int kbase = half * 32;
            v8f acc0 = {};
            v8f acc1 = {};
            #pragma unroll 4
            for (int i = 0; i < KITER; i += 2) {       // KITER = 65: odd, acc1 skipped on the last pass
                acc0 = wmma_chunk(aA, aB, brow, kbase + (i    ) * (KSPLIT * 32), inv, acc0);
                if (i + 1 < KITER)
                    acc1 = wmma_chunk(aA, aB, brow, kbase + (i + 1) * (KSPLIT * 32), inv, acc1);
            }
            if (lane < 16) {                 // D row 0: (M=0, N=lane) in acc[0]
                const int j = j0 + l15;
                if (j < NS) {
                    const float em = __expf(log_em[(size_t)j * NS + obs]);
                    const float y  = (acc0[0] + acc1[0]) * em;
                    wY[j] = y;               // single writer per (buffer, j)
                    atomicMax(wS, __float_as_uint(y));   // y >= 0: uint order == float order
                }
            }
        }
        grid_barrier(bar, bar + 1, (unsigned)(t + 1));
    }

    // Termination: out = L + log( sum_j y_T[j] * exp(log_trans[j][0]) )
    if (blockIdx.x == 0) {
        const float* fA = bufA[TLEN & 1];
        const float* fB = bufB[TLEN & 1];
        float part = 0.0f;
        for (int j = threadIdx.x; j < NS; j += BTHR)
            part += (fA[j] + fB[j]) * __expf(log_trans[(size_t)j * NS]);
        __shared__ float red[BTHR];
        red[threadIdx.x] = part;
        __syncthreads();
        for (int s = BTHR >> 1; s > 0; s >>= 1) {
            if (threadIdx.x < s) red[threadIdx.x] += red[threadIdx.x + s];
            __syncthreads();
        }
        if (threadIdx.x == 0) out[0] = (float)(L + log((double)red[0]));
    }
}

// ---------------------------------------------------------------------------
extern "C" void kernel_launch(void* const* d_in, const int* in_sizes, int n_in,
                              void* d_out, int out_size, void* d_ws, size_t ws_size,
                              hipStream_t stream) {
    const float* log_pi    = (const float*)d_in[0];
    const float* log_trans = (const float*)d_in[1];
    const float* log_em    = (const float*)d_in[2];
    const int*   obvs      = (const int*)d_in[3];
    float*       out       = (float*)d_out;

    char* ws = (char*)d_ws;
    size_t off = 0;
    _Float16* tT = (_Float16*)(ws + off);  off += (size_t)SP * SP * sizeof(_Float16); // ~34.6 MB
    float* yA0 = (float*)(ws + off);       off += (size_t)SP * sizeof(float);
    float* yA1 = (float*)(ws + off);       off += (size_t)SP * sizeof(float);
    float* yB0 = (float*)(ws + off);       off += (size_t)SP * sizeof(float);
    float* yB1 = (float*)(ws + off);       off += (size_t)SP * sizeof(float);
    unsigned* slotA = (unsigned*)(ws + off); off += (size_t)(TLEN + 4) * sizeof(unsigned);
    unsigned* slotB = (unsigned*)(ws + off); off += (size_t)(TLEN + 4) * sizeof(unsigned);
    unsigned* bar   = (unsigned*)(ws + off); off += 64;

    hmm_prep<<<dim3(SP / 32, SP / 32), dim3(32, 8), 0, stream>>>(log_trans, tT);
    hmm_init<<<(SP + 255) / 256, 256, 0, stream>>>(log_pi, yA0, yA1, yB0, yB1,
                                                   slotA, slotB, bar);
    hmm_forward<<<GBLK, BTHR, 0, stream>>>(tT, log_em, log_trans, obvs,
                                           yA0, yA1, yB0, yB1,
                                           slotA, slotB, bar, out);
}
